// FPAttention_82703890252561
// MI455X (gfx1250) — matrix operand order repaired and checked by reading, not compiled, not run on hardware
//
#include <hip/hip_runtime.h>

// ---------------------------------------------------------------------------
// FP Attention (Bahdanau) for MI455X / gfx1250, wave32 + WMMA bf16.
// B=32, T=4096, ENC=DEC=K=512.
// ---------------------------------------------------------------------------

#define BATCH 32
#define TLEN  4096
#define DIM   512          // ENC == DEC == K
#define WROW  1024         // W_attn row length (ENC+DEC)
#define AROW  520          // padded LDS row stride (bf16 elems): 512 + 8
#define MT    32           // t-rows per block (2 WMMA tiles, register-blocked)

typedef __attribute__((ext_vector_type(16))) __bf16 v16bf;
typedef __attribute__((ext_vector_type(8)))  __bf16 v8bf;
typedef __attribute__((ext_vector_type(8)))  float  v8f;
typedef __attribute__((ext_vector_type(2)))  __bf16 v2bf;

#if defined(__has_builtin) && __has_builtin(__builtin_amdgcn_tanhf)
#define TANHF(x) __builtin_amdgcn_tanhf(x)     // native v_tanh_f32
#else
#define TANHF(x) tanhf(x)
#endif

__device__ __forceinline__ unsigned short f2bf_bits(float f) {
    unsigned u = __float_as_uint(f);
    u = (u + 0x7FFFu + ((u >> 16) & 1u)) >> 16;     // round-to-nearest-even
    return (unsigned short)u;
}

// pack two floats into two bf16 in one dword
__device__ __forceinline__ unsigned pack2bf(float a, float b) {
#if defined(__has_builtin) && __has_builtin(__builtin_amdgcn_cvt_pk_bf16_f32)
    v2bf p = __builtin_amdgcn_cvt_pk_bf16_f32(a, b);
    unsigned r;
    __builtin_memcpy(&r, &p, 4);
    return r;
#else
    return (unsigned)f2bf_bits(a) | ((unsigned)f2bf_bits(b) << 16);
#endif
}

// ---------------------------------------------------------------------------
// proj_s[b,k] = sum_d s[b,d] * W_attn[k,d]   (one wave per (b,k))
// grid (64, 32) x 256 threads
// ---------------------------------------------------------------------------
__global__ __launch_bounds__(256) void projs_kernel(const float* __restrict__ s,
                                                    const float* __restrict__ W,
                                                    float* __restrict__ projs) {
    int b    = blockIdx.y;
    int k    = blockIdx.x * 8 + (threadIdx.x >> 5);
    int lane = threadIdx.x & 31;
    const float* wr = W + (size_t)k * WROW;          // W_s = W_attn[:, :512]
    const float* sr = s + (size_t)b * DIM;
    float acc = 0.f;
    #pragma unroll 4
    for (int d = lane; d < DIM; d += 32) acc += sr[d] * wr[d];
    #pragma unroll
    for (int off = 16; off; off >>= 1) acc += __shfl_xor(acc, off, 32);
    if (lane == 0) projs[(size_t)b * DIM + k] = acc;
}

// ---------------------------------------------------------------------------
// W_e (= W_attn[:, 512:]) -> bf16, row-major [k][e]  (pair-packed stores)
// grid 512 x 256
// ---------------------------------------------------------------------------
__global__ __launch_bounds__(256) void wecvt_kernel(const float* __restrict__ W,
                                                    unsigned* __restrict__ webf32) {
    int p = blockIdx.x * 256 + threadIdx.x;          // 0 .. 512*512/2-1
    int idx = p * 2;
    int k = idx >> 9, e = idx & 511;
    const float* src = W + (size_t)k * WROW + DIM + e;
    webf32[p] = pack2bf(src[0], src[1]);
}

// ---------------------------------------------------------------------------
// scores[b,t] = sum_k tanh(proj_s[b,k] + proj_e[b,t,k]) * v[k]
// proj_e computed on the fly with v_wmma_f32_16x16x32_bf16, never stored.
// Block = two 16-row t-tiles (A staged once in LDS as bf16); each B fragment
// is reused across both t-tiles (halves L2 traffic for W_e). 8 waves; wave w
// owns k-tiles (4w..4w+3)*16.   grid (T/32, B) x 256 threads
// ---------------------------------------------------------------------------
__global__ __launch_bounds__(256) void scores_kernel(const float* __restrict__ enc,
                                                     const float* __restrict__ projs,
                                                     const unsigned short* __restrict__ webf,
                                                     const float* __restrict__ v,
                                                     float* __restrict__ scores) {
    __shared__ __attribute__((aligned(16))) unsigned short abf[MT * AROW]; // 32.5 KB
    __shared__ float sc[MT];

    int b    = blockIdx.y;
    int t0   = blockIdx.x * MT;
    int tid  = threadIdx.x;
    int w    = tid >> 5;          // wave id 0..7
    int lane = tid & 31;
    int n    = lane & 15;         // column-in-tile (B/C/D) and row m (A)
    int h    = lane >> 4;         // half-wave select

    if (tid < MT) sc[tid] = 0.f;

    // ---- cooperative A-tile convert: enc[b, t0..t0+31, :] f32 -> bf16 in LDS
    {
        const float* base = enc + ((size_t)b * TLEN + t0) * DIM;
        #pragma unroll 8
        for (int it = 0; it < MT; it++) {            // 8192 pairs / 256 threads
            int p   = tid + it * 256;
            int idx = p * 2;
            int row = idx >> 9;
            int col = idx & 511;
            const float* sp = base + (size_t)row * DIM + col;
            *(unsigned*)&abf[row * AROW + col] = pack2bf(sp[0], sp[1]);
        }
    }
    __syncthreads();

    v8f acc[2][4];
    #pragma unroll
    for (int ti = 0; ti < 2; ti++)
        #pragma unroll
        for (int i = 0; i < 4; i++)
            #pragma unroll
            for (int r = 0; r < 8; r++) acc[ti][i][r] = 0.f;

    const unsigned short* arow0 = abf + n * AROW;          // A row m = n   (tile 0)
    const unsigned short* arow1 = abf + (n + 16) * AROW;   // A row m = n+16 (tile 1)

    for (int e0 = 0; e0 < DIM; e0 += 32) {
        // ---- A fragments (16x32 bf16): lane holds K = {16h..16h+15}
        v8bf lo0 = *(const v8bf*)(arow0 + e0 + 16 * h);        // ds_load_b128
        v8bf hi0 = *(const v8bf*)(arow0 + e0 + 16 * h + 8);
        v8bf lo1 = *(const v8bf*)(arow1 + e0 + 16 * h);
        v8bf hi1 = *(const v8bf*)(arow1 + e0 + 16 * h + 8);
        v16bf a0 = __builtin_shufflevector(lo0, hi0, 0, 1, 2, 3, 4, 5, 6, 7,
                                           8, 9, 10, 11, 12, 13, 14, 15);
        v16bf a1 = __builtin_shufflevector(lo1, hi1, 0, 1, 2, 3, 4, 5, 6, 7,
                                           8, 9, 10, 11, 12, 13, 14, 15);

        // ---- 4 k-tiles per wave; each B fragment reused for both t-tiles
        #pragma unroll
        for (int i = 0; i < 4; i++) {
            int k0 = (w * 4 + i) * 16;
            // B fragment (32x16 bf16): lane n = column, element j -> K = j + 16h.
            // Contiguous 16 bf16 (32B aligned) => two global_load_b128.
            const v16bf* bp = (const v16bf*)(webf + ((size_t)(k0 + n)) * DIM + e0 + 16 * h);
            v16bf bfrag = *bp;
            acc[0][i] = __builtin_amdgcn_wmma_f32_16x16x32_bf16(
                false, a0, false, bfrag, (short)0, acc[0][i], false, false);
            acc[1][i] = __builtin_amdgcn_wmma_f32_16x16x32_bf16(
                false, a1, false, bfrag, (short)0, acc[1][i], false, false);
        }
    }

    // ---- fused epilogue: tanh(proj_s + proj_e) * v, reduce over k-columns
    const float* psb = projs + (size_t)b * DIM;
    #pragma unroll
    for (int i = 0; i < 4; i++) {
        int   col = (w * 4 + i) * 16 + n;
        float ps  = psb[col];
        float vv  = v[col];
        #pragma unroll
        for (int ti = 0; ti < 2; ti++) {
            #pragma unroll
            for (int r = 0; r < 8; r++) {
                // acc[ti][i][r] = proj_e[t0 + 16*ti + r + 8h, col]
                float val = TANHF(ps + acc[ti][i][r]) * vv;
                #pragma unroll
                for (int off = 1; off < 16; off <<= 1)   // reduce over 16 lanes of half
                    val += __shfl_xor(val, off, 32);
                if (n == 0) atomicAdd(&sc[16 * ti + r + 8 * h], val);
            }
        }
    }
    __syncthreads();
    if (tid < MT) scores[(size_t)b * TLEN + t0 + tid] = sc[tid];
}

// ---------------------------------------------------------------------------
// In-place softmax over T per batch. grid 32 x 256, 16 elems/thread.
// ---------------------------------------------------------------------------
__global__ __launch_bounds__(256) void softmax_kernel(float* __restrict__ scores) {
    __shared__ float red[8];
    int b   = blockIdx.x;
    int tid = threadIdx.x;
    float* sc = scores + (size_t)b * TLEN;

    float vals[16];
    float m = -3.4e38f;
    #pragma unroll
    for (int j = 0; j < 16; j++) { vals[j] = sc[tid + j * 256]; m = fmaxf(m, vals[j]); }
    #pragma unroll
    for (int off = 16; off; off >>= 1) m = fmaxf(m, __shfl_xor(m, off, 32));
    if ((tid & 31) == 0) red[tid >> 5] = m;
    __syncthreads();
    if (tid == 0) {
        float mm = red[0];
        for (int i = 1; i < 8; i++) mm = fmaxf(mm, red[i]);
        red[0] = mm;
    }
    __syncthreads();
    m = red[0];
    __syncthreads();

    float s = 0.f;
    #pragma unroll
    for (int j = 0; j < 16; j++) { vals[j] = __expf(vals[j] - m); s += vals[j]; }
    #pragma unroll
    for (int off = 16; off; off >>= 1) s += __shfl_xor(s, off, 32);
    if ((tid & 31) == 0) red[tid >> 5] = s;
    __syncthreads();
    if (tid == 0) {
        float ss = 0.f;
        for (int i = 0; i < 8; i++) ss += red[i];
        red[0] = ss;
    }
    __syncthreads();
    float inv = 1.f / red[0];
    #pragma unroll
    for (int j = 0; j < 16; j++) sc[tid + j * 256] = vals[j] * inv;
}

// ---------------------------------------------------------------------------
// c partials: cpart[b, tc, e] = sum over 512 t's of att * enc  (coalesced on e)
// grid (8, 32) x 512
// ---------------------------------------------------------------------------
__global__ __launch_bounds__(512) void ctx_partial_kernel(const float* __restrict__ enc,
                                                          const float* __restrict__ att,
                                                          float* __restrict__ cpart) {
    int b  = blockIdx.y;
    int tc = blockIdx.x;                 // 0..7 (512 timesteps each)
    int e  = threadIdx.x;                // 0..511
    const float* a  = att + (size_t)b * TLEN + tc * 512;
    const float* ep = enc + ((size_t)b * TLEN + (size_t)tc * 512) * DIM + e;
    float acc = 0.f;
    #pragma unroll 4
    for (int t = 0; t < 512; t++) acc += a[t] * ep[(size_t)t * DIM];
    cpart[((size_t)b * 8 + tc) * DIM + e] = acc;
}

// grid 32 x 512
__global__ __launch_bounds__(512) void ctx_reduce_kernel(const float* __restrict__ cpart,
                                                         float* __restrict__ out) {
    int b = blockIdx.x;
    int e = threadIdx.x;
    float acc = 0.f;
    #pragma unroll
    for (int i = 0; i < 8; i++) acc += cpart[((size_t)b * 8 + i) * DIM + e];
    out[(size_t)b * DIM + e] = acc;
}

// ---------------------------------------------------------------------------
// Launch. Inputs: s(32x512) f32, enc(32x4096x512) f32, W_attn(512x1024) f32,
// v(1x512) f32. Output c: 32x1x512 f32.
// ws layout (bytes):
//   [0,        65536)   proj_s         32*512 f32
//   [65536,   589824)   W_e bf16       512*512 u16
//   [589824, 1114112)   scores/att     32*4096 f32
//   [1114112,1638400)   c partials     32*8*512 f32
// ---------------------------------------------------------------------------
extern "C" void kernel_launch(void* const* d_in, const int* in_sizes, int n_in,
                              void* d_out, int out_size, void* d_ws, size_t ws_size,
                              hipStream_t stream) {
    const float* s   = (const float*)d_in[0];
    const float* enc = (const float*)d_in[1];
    const float* W   = (const float*)d_in[2];
    const float* v   = (const float*)d_in[3];
    float* out = (float*)d_out;

    char* ws = (char*)d_ws;
    float*          projs  = (float*)(ws + 0);
    unsigned short* webf   = (unsigned short*)(ws + 65536);
    float*          scores = (float*)(ws + 589824);
    float*          cpart  = (float*)(ws + 1114112);

    projs_kernel<<<dim3(64, BATCH), 256, 0, stream>>>(s, W, projs);
    wecvt_kernel<<<dim3((DIM * DIM / 2) / 256), 256, 0, stream>>>(W, (unsigned*)webf);
    scores_kernel<<<dim3(TLEN / MT, BATCH), 256, 0, stream>>>(enc, projs, webf, v, scores);
    softmax_kernel<<<dim3(BATCH), 256, 0, stream>>>(scores);
    ctx_partial_kernel<<<dim3(8, BATCH), 512, 0, stream>>>(enc, scores, cpart);
    ctx_reduce_kernel<<<dim3(BATCH), 512, 0, stream>>>(cpart, out);
}
